// CorticalRecurrentStack_13735305413042
// MI455X (gfx1250) — matrix operand order repaired and checked by reading, not compiled, not run on hardware
//
#include <hip/hip_runtime.h>
#include <hip/hip_bf16.h>
#include <math.h>

// Problem constants (match reference)
#define LNUM 4
#define TT   512
#define BB   32
#define DD   1024
#define EPSV 1e-5f

// Persistent recurrent kernel config: 32 blocks x 128 threads = 128 waves
// = exactly one wave per 16x16 tile of the (32 x 1024) per-layer output.
#define NBLK 32
#define PTHR 128

typedef __bf16 bf16;
typedef __attribute__((ext_vector_type(16))) bf16  v16bf;
typedef __attribute__((ext_vector_type(8)))  bf16  v8bf;
typedef __attribute__((ext_vector_type(8)))  float v8f;

// ---------------------------------------------------------------------------
// WMMA helpers (A: 16x32 bf16 row-major source; B: weights pre-transposed to
// [N][K] so each lane's 16 K-values are 32 contiguous bytes -> global_load_b128)
// ---------------------------------------------------------------------------
__device__ __forceinline__ v16bf load_a_tile(const bf16* __restrict__ arow, int k, int lh) {
    // A 16x32 bf16 layout: lane<16 -> M=lane, K = k..k+15
    //                      lane>=16 -> M=lane-16, K = k+16..k+31
    const bf16* p = arow + k + lh * 16;
    v8bf lo = *(const v8bf*)(p);
    v8bf hi = *(const v8bf*)(p + 8);
    return __builtin_shufflevector(lo, hi, 0,1,2,3,4,5,6,7,8,9,10,11,12,13,14,15);
}

__device__ __forceinline__ v16bf load_b_tile(const bf16* __restrict__ bnrow, int k, int lh) {
    // B 32x16 bf16 layout: lane&15 -> N, lane>>4 selects K half; weights stored [N][K]
    return *(const v16bf*)(bnrow + k + lh * 16);
}

// Accumulate accf += A@Bf, accc += A@Bc over K = DD (shared A loads for both gates)
__device__ __forceinline__ void gemm_k_loop(v8f& accf, v8f& accc,
                                            const bf16* __restrict__ arow,
                                            const bf16* __restrict__ bf_row,
                                            const bf16* __restrict__ bc_row,
                                            int lh) {
#pragma unroll 4
    for (int k = 0; k < DD; k += 32) {
        v16bf a  = load_a_tile(arow, k, lh);
        v16bf bf_ = load_b_tile(bf_row, k, lh);
        v16bf bc_ = load_b_tile(bc_row, k, lh);
        accf = __builtin_amdgcn_wmma_f32_16x16x32_bf16(false, a, false, bf_, (short)0, accf, false, false);
        accc = __builtin_amdgcn_wmma_f32_16x16x32_bf16(false, a, false, bc_, (short)0, accc, false, false);
    }
}

// ---------------------------------------------------------------------------
// Block reduction (sum, sumsq) via LDS
// ---------------------------------------------------------------------------
__device__ __forceinline__ float2 block_reduce2(float a, float b, float* r1, float* r2, int nthr) {
    int t = threadIdx.x;
    r1[t] = a; r2[t] = b;
    __syncthreads();
    for (int s = nthr >> 1; s > 0; s >>= 1) {
        if (t < s) { r1[t] += r1[t + s]; r2[t] += r2[t + s]; }
        __syncthreads();
    }
    float2 out = make_float2(r1[0], r2[0]);
    __syncthreads();
    return out;
}

// ---------------------------------------------------------------------------
// Device-scope grid barrier (monotone counter, sense-free: target = epoch*NBLK)
// ---------------------------------------------------------------------------
__device__ __forceinline__ void grid_barrier(unsigned* cnt, unsigned target) {
    __threadfence();           // make my writes device-visible
    __syncthreads();
    if (threadIdx.x == 0) {
        atomicAdd(cnt, 1u);
        while (atomicAdd(cnt, 0u) < target) { __builtin_amdgcn_s_sleep(1); }
    }
    __syncthreads();
    // Drop stale WGP$ lines before reading other blocks' data
    asm volatile("global_inv scope:SCOPE_DEV" ::: "memory");
    __threadfence();
}

// ---------------------------------------------------------------------------
// K1: convert weights f32 [L][K][N] -> bf16 transposed [L][N][K]
// ---------------------------------------------------------------------------
__global__ void __launch_bounds__(256) convert_weights_kernel(
    const float* __restrict__ Wf, const float* __restrict__ Uf,
    const float* __restrict__ Wc, const float* __restrict__ Uc,
    bf16* __restrict__ Wtf, bf16* __restrict__ Utf,
    bf16* __restrict__ Wtc, bf16* __restrict__ Utc) {
    const size_t total = (size_t)LNUM * DD * DD;
    for (size_t o = (size_t)blockIdx.x * blockDim.x + threadIdx.x; o < total;
         o += (size_t)gridDim.x * blockDim.x) {
        size_t l   = o / ((size_t)DD * DD);
        size_t rem = o % ((size_t)DD * DD);
        size_t n   = rem / DD;
        size_t k   = rem % DD;
        size_t i   = l * DD * DD + k * DD + n;
        Wtf[o] = (bf16)Wf[i];
        Utf[o] = (bf16)Uf[i];
        Wtc[o] = (bf16)Wc[i];
        Utc[o] = (bf16)Uc[i];
    }
}

// ---------------------------------------------------------------------------
// K2: init state: f32 copy into ws + bf16 mirror (A-matrix for U-path WMMA)
// ---------------------------------------------------------------------------
__global__ void __launch_bounds__(256) init_state_kernel(
    const float* __restrict__ st, float* __restrict__ s, bf16* __restrict__ sbf) {
    const size_t total = (size_t)LNUM * BB * DD;
    for (size_t o = (size_t)blockIdx.x * blockDim.x + threadIdx.x; o < total;
         o += (size_t)gridDim.x * blockDim.x) {
        float v = st[o];
        s[o]   = v;
        sbf[o] = (bf16)v;
    }
}

// ---------------------------------------------------------------------------
// K3: LayerNorm(x_t) for ALL t with layer-0 pre-norm params -> xn bf16.
// One block per (t,b) row (T*B = 16384 rows), 256 threads reduce D=1024.
// ---------------------------------------------------------------------------
__global__ void __launch_bounds__(256) ln_x_kernel(
    const float* __restrict__ x, const float* __restrict__ g0,
    const float* __restrict__ b0, bf16* __restrict__ xn) {
    __shared__ float r1[256];
    __shared__ float r2[256];
    const int row = blockIdx.x;
    const float* xr = x + (size_t)row * DD;
    float s = 0.f, ss = 0.f;
    for (int e = threadIdx.x; e < DD; e += 256) { float v = xr[e]; s += v; ss += v * v; }
    float2 tot = block_reduce2(s, ss, r1, r2, 256);
    const float m    = tot.x * (1.f / DD);
    const float rstd = rsqrtf(tot.y * (1.f / DD) - m * m + EPSV);
    for (int e = threadIdx.x; e < DD; e += 256)
        xn[(size_t)row * DD + e] = (bf16)((xr[e] - m) * rstd * g0[e] + b0[e]);
}

// ---------------------------------------------------------------------------
// K4: time-parallel precompute of layer-0 input path:
//   xWf = LN(x)@Wf0 + bf0 ; xWc = LN(x)@Wc0 + bc0   (16384 x 1024, K=1024)
// 65536 waves, one 16x16 tile each, 64 WMMAs per wave.
// ---------------------------------------------------------------------------
__global__ void __launch_bounds__(256) precompute_xw_kernel(
    const bf16* __restrict__ xn, const bf16* __restrict__ Wtf, const bf16* __restrict__ Wtc,
    const float* __restrict__ bfb, const float* __restrict__ bcb,
    float* __restrict__ xWf, float* __restrict__ xWc) {
    const int lane  = threadIdx.x & 31;
    const int wid   = blockIdx.x * 8 + (threadIdx.x >> 5);
    const int mtile = wid >> 6;   // 0..1023  (rows of T*B)
    const int ntile = wid & 63;   // 0..63    (cols of D)
    const int l15 = lane & 15, lh = lane >> 4;
    const bf16* arow = xn  + (size_t)(mtile * 16 + l15) * DD;
    const bf16* bfr  = Wtf + (size_t)(ntile * 16 + l15) * DD;
    const bf16* bcr  = Wtc + (size_t)(ntile * 16 + l15) * DD;
    const int gn = ntile * 16 + l15;
    const float bfv = bfb[gn], bcv = bcb[gn];
    v8f accf = {}, accc = {};
#pragma unroll
    for (int j = 0; j < 8; ++j) { accf[j] = bfv; accc[j] = bcv; }
    gemm_k_loop(accf, accc, arow, bfr, bcr, lh);
#pragma unroll
    for (int j = 0; j < 8; ++j) {
        int gm = mtile * 16 + j + lh * 8;
        xWf[(size_t)gm * DD + gn] = accf[j];
        xWc[(size_t)gm * DD + gn] = accc[j];
    }
}

// ---------------------------------------------------------------------------
// LN of one B-row (row = blockIdx.x) inside the persistent kernel
// ---------------------------------------------------------------------------
__device__ __forceinline__ void ln_row_to_bf16(const float* __restrict__ src, bf16* __restrict__ dst,
                                               const float* __restrict__ g, const float* __restrict__ b,
                                               float* r1, float* r2) {
    const int row = blockIdx.x;
    const float* xr = src + (size_t)row * DD;
    float s = 0.f, ss = 0.f;
    for (int e = threadIdx.x; e < DD; e += PTHR) { float v = xr[e]; s += v; ss += v * v; }
    float2 tot = block_reduce2(s, ss, r1, r2, PTHR);
    const float m    = tot.x * (1.f / DD);
    const float rstd = rsqrtf(tot.y * (1.f / DD) - m * m + EPSV);
    for (int e = threadIdx.x; e < DD; e += PTHR)
        dst[(size_t)row * DD + e] = (bf16)((xr[e] - m) * rstd * g[e] + b[e]);
}

__device__ __forceinline__ void ln_row_to_f32(const float* __restrict__ src, float* __restrict__ dst,
                                              const float* __restrict__ g, const float* __restrict__ b,
                                              float* r1, float* r2) {
    const int row = blockIdx.x;
    const float* xr = src + (size_t)row * DD;
    float s = 0.f, ss = 0.f;
    for (int e = threadIdx.x; e < DD; e += PTHR) { float v = xr[e]; s += v; ss += v * v; }
    float2 tot = block_reduce2(s, ss, r1, r2, PTHR);
    const float m    = tot.x * (1.f / DD);
    const float rstd = rsqrtf(tot.y * (1.f / DD) - m * m + EPSV);
    for (int e = threadIdx.x; e < DD; e += PTHR)
        dst[(size_t)row * DD + e] = (xr[e] - m) * rstd * g[e] + b[e];
}

// ---------------------------------------------------------------------------
// K5: persistent recurrent kernel. 32 blocks x 128 threads = 128 waves.
// Wave w -> output tile (mtile = w>>6 in {0,1}, ntile = w&63). Per layer phase:
// up to 128 v_wmma_f32_16x16x32_bf16 per wave, then gate fusion in registers.
// ---------------------------------------------------------------------------
__global__ void __launch_bounds__(PTHR) recurrent_kernel(
    const float* __restrict__ xWf, const float* __restrict__ xWc,
    const bf16* __restrict__ Wtf, const bf16* __restrict__ Utf,
    const bf16* __restrict__ Wtc, const bf16* __restrict__ Utc,
    const float* __restrict__ bfb, const float* __restrict__ bcb,
    const float* __restrict__ pre_g, const float* __restrict__ pre_b,
    const float* __restrict__ st_g, const float* __restrict__ st_b,
    float* __restrict__ s, bf16* __restrict__ sbf,
    float* __restrict__ h, bf16* __restrict__ hn,
    float* __restrict__ y, float* __restrict__ state_out,
    unsigned* __restrict__ barcnt) {
    __shared__ float r1[PTHR];
    __shared__ float r2[PTHR];
    const int lane  = threadIdx.x & 31;
    const int wid   = blockIdx.x * (PTHR / 32) + (threadIdx.x >> 5);  // 0..127
    const int mtile = wid >> 6;                                       // 0..1
    const int ntile = wid & 63;                                       // 0..63
    const int l15 = lane & 15, lh = lane >> 4;
    const int gn = ntile * 16 + l15;
    unsigned epoch = 0;

    for (int t = 0; t < TT; ++t) {
        for (int i = 0; i < LNUM; ++i) {
            if (i > 0) {
                // pre-norm of current h (layer 0's LN + W-path was precomputed)
                ln_row_to_bf16(h, hn, pre_g + (size_t)i * DD, pre_b + (size_t)i * DD, r1, r2);
                ++epoch; grid_barrier(barcnt, epoch * NBLK);
            }
            v8f accf = {}, accc = {};
            if (i == 0) {
                // accumulator seeded with precomputed LN(x_t)@W + bias
#pragma unroll
                for (int j = 0; j < 8; ++j) {
                    int gm = mtile * 16 + j + lh * 8;
                    size_t r = ((size_t)t * BB + gm) * DD + gn;
                    accf[j] = xWf[r];
                    accc[j] = xWc[r];
                }
            } else {
                const float bfv = bfb[(size_t)i * DD + gn];
                const float bcv = bcb[(size_t)i * DD + gn];
#pragma unroll
                for (int j = 0; j < 8; ++j) { accf[j] = bfv; accc[j] = bcv; }
                gemm_k_loop(accf, accc,
                            hn  + (size_t)(mtile * 16 + l15) * DD,
                            Wtf + ((size_t)i * DD + ntile * 16 + l15) * DD,
                            Wtc + ((size_t)i * DD + ntile * 16 + l15) * DD, lh);
            }
            // state (U) path: s_{t-1}[i] @ Uf/Uc
            gemm_k_loop(accf, accc,
                        sbf + ((size_t)i * BB + mtile * 16 + l15) * DD,
                        Utf + ((size_t)i * DD + ntile * 16 + l15) * DD,
                        Utc + ((size_t)i * DD + ntile * 16 + l15) * DD, lh);
            // gate fusion: ns = sigmoid(f)*s + (1-sigmoid(f))*tanh(c)
#pragma unroll
            for (int j = 0; j < 8; ++j) {
                int gm = mtile * 16 + j + lh * 8;
                size_t sidx = (size_t)i * BB * DD + (size_t)gm * DD + gn;
                float fg = 1.f / (1.f + __expf(-accf[j]));
                float cd = tanhf(accc[j]);
                float sv = s[sidx];
                float ns = fg * sv + (1.f - fg) * cd;
                s[sidx]   = ns;
                sbf[sidx] = (bf16)ns;
                h[(size_t)gm * DD + gn] = ns;
            }
            ++epoch; grid_barrier(barcnt, epoch * NBLK);
        }
        // stack output norm -> y_t (f32)
        ln_row_to_f32(h, y + (size_t)t * BB * DD, st_g, st_b, r1, r2);
        ++epoch; grid_barrier(barcnt, epoch * NBLK);
    }
    // emit final states
    for (size_t idx = (size_t)blockIdx.x * PTHR + threadIdx.x; idx < (size_t)LNUM * BB * DD;
         idx += (size_t)NBLK * PTHR)
        state_out[idx] = s[idx];
}

// ---------------------------------------------------------------------------
// Host side
// ---------------------------------------------------------------------------
extern "C" void kernel_launch(void* const* d_in, const int* in_sizes, int n_in,
                              void* d_out, int out_size, void* d_ws, size_t ws_size,
                              hipStream_t stream) {
    const float* x     = (const float*)d_in[0];   // (T,B,D)
    const float* state = (const float*)d_in[1];   // (L,B,D)
    const float* Wf    = (const float*)d_in[2];   // (L,D,D)
    const float* Uf    = (const float*)d_in[3];
    const float* bfb   = (const float*)d_in[4];   // (L,D)
    const float* Wc    = (const float*)d_in[5];
    const float* Uc    = (const float*)d_in[6];
    const float* bcb   = (const float*)d_in[7];
    const float* pre_g = (const float*)d_in[8];   // (L,D)
    const float* pre_b = (const float*)d_in[9];
    const float* st_g  = (const float*)d_in[10];  // (D)
    const float* st_b  = (const float*)d_in[11];

    float* y_out     = (float*)d_out;                          // (T,B,D)
    float* state_out = (float*)d_out + (size_t)TT * BB * DD;   // (L,B,D)

    // Workspace carve-up (256B aligned)
    char*  base = (char*)d_ws;
    size_t off  = 0;
    auto carve = [&](size_t bytes) -> char* {
        char* p = base + off;
        off += (bytes + 255) & ~(size_t)255;
        return p;
    };
    unsigned* barcnt = (unsigned*)carve(256);
    bf16* Wtf = (bf16*)carve((size_t)LNUM * DD * DD * sizeof(bf16));
    bf16* Utf = (bf16*)carve((size_t)LNUM * DD * DD * sizeof(bf16));
    bf16* Wtc = (bf16*)carve((size_t)LNUM * DD * DD * sizeof(bf16));
    bf16* Utc = (bf16*)carve((size_t)LNUM * DD * DD * sizeof(bf16));
    bf16* xn  = (bf16*)carve((size_t)TT * BB * DD * sizeof(bf16));
    float* xWf = (float*)carve((size_t)TT * BB * DD * sizeof(float));
    float* xWc = (float*)carve((size_t)TT * BB * DD * sizeof(float));
    float* s   = (float*)carve((size_t)LNUM * BB * DD * sizeof(float));
    bf16*  sbfp = (bf16*)carve((size_t)LNUM * BB * DD * sizeof(bf16));
    float* h   = (float*)carve((size_t)BB * DD * sizeof(float));
    bf16*  hn  = (bf16*)carve((size_t)BB * DD * sizeof(bf16));

    // barrier counter must be zero each launch (graph replay safe)
    hipMemsetAsync(barcnt, 0, 256, stream);

    convert_weights_kernel<<<4096, 256, 0, stream>>>(Wf, Uf, Wc, Uc, Wtf, Utf, Wtc, Utc);
    init_state_kernel<<<256, 256, 0, stream>>>(state, s, sbfp);
    ln_x_kernel<<<TT * BB, 256, 0, stream>>>(x, pre_g, pre_b, xn);  // layer-0 params are rows 0
    precompute_xw_kernel<<<8192, 256, 0, stream>>>(xn, Wtf, Wtc, bfb, bcb, xWf, xWc);
    recurrent_kernel<<<NBLK, PTHR, 0, stream>>>(xWf, xWc, Wtf, Utf, Wtc, Utc,
                                                bfb, bcb, pre_g, pre_b, st_g, st_b,
                                                s, sbfp, h, hn, y_out, state_out, barcnt);
    (void)in_sizes; (void)n_in; (void)out_size; (void)ws_size;
}
